// WanSelfAttention_64836826301140
// MI455X (gfx1250) — matrix operand order repaired and checked by reading, not compiled, not run on hardware
//
#include <hip/hip_runtime.h>

#define DIM 1536
#define SEQ 3072
#define NH 12
#define HD 128
#define C2 64

typedef __attribute__((ext_vector_type(16))) __bf16 v16bf;
typedef __attribute__((ext_vector_type(8)))  __bf16 v8bf;
typedef __attribute__((ext_vector_type(8)))  float  v8f;

__device__ inline v8f wmma_bf16(v16bf a, v16bf b, v8f c) {
    return __builtin_amdgcn_wmma_f32_16x16x32_bf16(
        /*neg_a=*/false, a, /*neg_b=*/false, b,
        /*c_mod=*/(short)0, c, /*reuse_a=*/false, /*reuse_b=*/false);
}

__device__ inline v16bf cat8(v8bf lo, v8bf hi) {
    return __builtin_shufflevector(lo, hi, 0,1,2,3,4,5,6,7,8,9,10,11,12,13,14,15);
}

// Async 16B global->LDS copy (per lane), tracked by ASYNCcnt.
__device__ inline void async_copy16(const void* gsrc, void* ldst) {
    unsigned lds_addr = (unsigned)(size_t)ldst;   // flat LDS addr low bits = LDS offset
    asm volatile("global_load_async_to_lds_b128 %0, %1, off"
                 :: "v"(lds_addr), "v"(gsrc) : "memory");
}
__device__ inline void wait_async0() {
    asm volatile("s_wait_asynccnt 0" ::: "memory");
}

// ---------------------------------------------------------------- convert
__global__ void cvt_f32_bf16(const float* __restrict__ in,
                             __bf16* __restrict__ out, int n) {
    int i = blockIdx.x * blockDim.x + threadIdx.x;
    if (i < n) out[i] = (__bf16)in[i];
}

// f32 [R][C] -> bf16 [C][R] (tiled transpose through LDS)
__global__ __launch_bounds__(256)
void cvt_transpose(const float* __restrict__ in, __bf16* __restrict__ out,
                   int R, int C) {
    __shared__ float tile[32][33];
    int rb = blockIdx.y * 32, cb = blockIdx.x * 32;
    int tx = threadIdx.x & 31, ty = threadIdx.x >> 5;   // 8 row-groups
#pragma unroll
    for (int i = ty; i < 32; i += 8)
        tile[i][tx] = in[(size_t)(rb + i) * C + cb + tx];
    __syncthreads();
#pragma unroll
    for (int i = ty; i < 32; i += 8)
        out[(size_t)(cb + i) * R + rb + tx] = (__bf16)tile[tx][i];
}

// ---------------------------------------------------------------- GEMM
// C[M,N] = A[M,K](bf16) * Bt[N,K](bf16, pre-transposed) + bias[N], f32 out.
// Block: 256 thr (8 waves). Block tile 128x128, wave tile 64x32, BK=32.
#define BM 128
#define BN 128
#define BK 32
#define LDT 40   // padded LDS stride (elements); 80B rows, 16B aligned

__global__ __launch_bounds__(256)
void gemm_bf16(const __bf16* __restrict__ A, const __bf16* __restrict__ Bt,
               const float* __restrict__ bias, float* __restrict__ C,
               int M, int N, int K) {
    __shared__ __bf16 As[2][BM * LDT];
    __shared__ __bf16 Bs[2][BN * LDT];

    const int tid  = threadIdx.x;
    const int lane = tid & 31;
    const int wave = tid >> 5;       // 0..7
    const int wm   = wave & 1;       // 2 waves along M
    const int wn   = wave >> 1;      // 4 waves along N
    const int lr   = lane & 15;
    const int hi2  = lane >> 4;      // 0/1

    const int mBlock = blockIdx.y * BM;
    const int nBlock = blockIdx.x * BN;

    auto stageA = [&](int buf, int k0) {
#pragma unroll
        for (int i = 0; i < 2; ++i) {
            int idx = tid + i * 256;           // 0..511, 8 elems each
            int row = idx >> 2;                // 0..127
            int cg  = idx & 3;                 // 0..3
            async_copy16(A + (size_t)(mBlock + row) * K + k0 + cg * 8,
                         &As[buf][row * LDT + cg * 8]);
        }
    };
    auto stageB = [&](int buf, int k0) {
#pragma unroll
        for (int i = 0; i < 2; ++i) {
            int idx = tid + i * 256;
            int row = idx >> 2;                // 0..127 (N rows of Bt)
            int cg  = idx & 3;
            async_copy16(Bt + (size_t)(nBlock + row) * K + k0 + cg * 8,
                         &Bs[buf][row * LDT + cg * 8]);
        }
    };

    v8f acc[4][2];
#pragma unroll
    for (int m = 0; m < 4; ++m)
#pragma unroll
        for (int n = 0; n < 2; ++n) acc[m][n] = (v8f)0.0f;

    stageA(0, 0); stageB(0, 0);
    wait_async0();
    __syncthreads();

    const int ksteps = K / BK;
    for (int kt = 0; kt < ksteps; ++kt) {
        int cur = kt & 1;
        if (kt + 1 < ksteps) { stageA(cur ^ 1, (kt + 1) * BK); stageB(cur ^ 1, (kt + 1) * BK); }

        v16bf af[4], bfr[2];
#pragma unroll
        for (int m = 0; m < 4; ++m) {
            const __bf16* p = &As[cur][(wm * 64 + m * 16 + lr) * LDT];
            v8bf lo = *(const v8bf*)(p + hi2 * 16);
            v8bf hb = *(const v8bf*)(p + hi2 * 16 + 8);
            af[m] = cat8(lo, hb);
        }
#pragma unroll
        for (int n = 0; n < 2; ++n) {
            const __bf16* p = &Bs[cur][(wn * 32 + n * 16 + lr) * LDT + hi2 * 16];
            v8bf lo = *(const v8bf*)p;
            v8bf hb = *(const v8bf*)(p + 8);
            bfr[n] = cat8(lo, hb);
        }
#pragma unroll
        for (int m = 0; m < 4; ++m)
#pragma unroll
            for (int n = 0; n < 2; ++n)
                acc[m][n] = wmma_bf16(af[m], bfr[n], acc[m][n]);

        wait_async0();
        __syncthreads();
    }

#pragma unroll
    for (int m = 0; m < 4; ++m)
#pragma unroll
        for (int n = 0; n < 2; ++n) {
            int gn = nBlock + wn * 32 + n * 16 + lr;
            float bv = bias ? bias[gn] : 0.0f;
            int gm0 = mBlock + wm * 64 + m * 16 + hi2 * 8;
#pragma unroll
            for (int r = 0; r < 8; ++r)
                C[(size_t)(gm0 + r) * N + gn] = acc[m][n][r] + bv;
        }
}

// ---------------------------------------------------------------- RMSNorm + RoPE
__global__ __launch_bounds__(256)
void rms_rope(const float* __restrict__ q0, const float* __restrict__ k0,
              const float* __restrict__ freqs,
              const float* __restrict__ gq, const float* __restrict__ gk,
              __bf16* __restrict__ qb, __bf16* __restrict__ kb) {
    int s = blockIdx.x;
    int t = threadIdx.x;
    int lane = t & 31, wave = t >> 5;
    const float* qr = q0 + (size_t)s * DIM;
    const float* kr = k0 + (size_t)s * DIM;

    __shared__ float redq[8], redk[8];
    float sq = 0.f, sk = 0.f;
    for (int c = t; c < DIM; c += 256) {
        float a = qr[c]; sq += a * a;
        float b = kr[c]; sk += b * b;
    }
#pragma unroll
    for (int off = 16; off > 0; off >>= 1) {
        sq += __shfl_xor(sq, off, 32);
        sk += __shfl_xor(sk, off, 32);
    }
    if (lane == 0) { redq[wave] = sq; redk[wave] = sk; }
    __syncthreads();
    float tq = 0.f, tk = 0.f;
#pragma unroll
    for (int i = 0; i < 8; ++i) { tq += redq[i]; tk += redk[i]; }
    float rq = rsqrtf(tq / (float)DIM + 1e-6f);
    float rk = rsqrtf(tk / (float)DIM + 1e-6f);

    int fidx = s >> 10;          // / (32*32)
    int hidx = (s >> 5) & 31;
    int widx = s & 31;

    for (int p = t; p < NH * C2; p += 256) {   // 768 rope pairs
        int c2 = p & 63;
        int pos = (c2 < 22) ? fidx : ((c2 < 43) ? hidx : widx);
        float ang = freqs[pos * 64 + c2];
        float sn, cs;
        __sincosf(ang, &sn, &cs);
        int ch = (p >> 6) * HD + c2 * 2;
        size_t base = (size_t)s * DIM + ch;
        float tr = q0[base]     * rq * gq[ch];
        float ti = q0[base + 1] * rq * gq[ch + 1];
        qb[base]     = (__bf16)(tr * cs - ti * sn);
        qb[base + 1] = (__bf16)(tr * sn + ti * cs);
        tr = k0[base]     * rk * gk[ch];
        ti = k0[base + 1] * rk * gk[ch + 1];
        kb[base]     = (__bf16)(tr * cs - ti * sn);
        kb[base + 1] = (__bf16)(tr * sn + ti * cs);
    }
}

// ---------------------------------------------------------------- Flash attention
// One block = (head, 16-query rows), 256 threads = 8 waves.
// vt is V^T: bf16 [DIM][SEQ]  (row = channel h*HD+d, col = key)
#define KCHUNK 512
#define SCS 516          // padded f32 stride of score rows
#define VLD 40           // padded bf16 stride of vT rows

__global__ __launch_bounds__(256)
void flash_attn(const __bf16* __restrict__ qb, const __bf16* __restrict__ kb,
                const __bf16* __restrict__ vt, const int* __restrict__ seq_lens,
                __bf16* __restrict__ attnb) {
    __shared__ float  sc[16 * SCS];        // 33 KB
    __shared__ __bf16 vT[2][HD * VLD];     // 20 KB
    __shared__ float  fac[16];
    __shared__ float  inv_s[16];

    const int qbase = blockIdx.x * 16;
    const int h     = blockIdx.y;
    const int tid = threadIdx.x, lane = tid & 31, wave = tid >> 5;
    const int lr = lane & 15, hi2 = lane >> 4;
    const int slen = seq_lens[0];
    const float scale = 0.08838834764831845f;   // 128^-0.5

    // Q fragments (16x32 x 4 along head dim), loaded once from global.
    v16bf qf[4];
    {
        const __bf16* qrow = qb + (size_t)(qbase + lr) * DIM + h * HD;
#pragma unroll
        for (int c = 0; c < 4; ++c) {
            v8bf lo = *(const v8bf*)(qrow + c * 32 + hi2 * 16);
            v8bf hb = *(const v8bf*)(qrow + c * 32 + hi2 * 16 + 8);
            qf[c] = cat8(lo, hb);
        }
    }

    // Stage 32-key slab of V^T (128 dims x 32 keys) into LDS via async copies.
    auto stageV = [&](int buf, int key0) {
#pragma unroll
        for (int i = 0; i < 2; ++i) {
            int idx = tid + i * 256;       // 0..511
            int d   = idx >> 2;            // 0..127 (head channel)
            int sub = idx & 3;             // key group of 8
            async_copy16(vt + (size_t)(h * HD + d) * SEQ + key0 + sub * 8,
                         &vT[buf][d * VLD + sub * 8]);
        }
    };

    float m_run = -3.0e38f, s_run = 0.f;   // per (tid>>4) query row
    const int srow_id = tid >> 4;          // 0..15
    const int sl      = tid & 15;
    v8f oacc = (v8f)0.0f;                  // wave's 16x16 output slice

    for (int chunk0 = 0; chunk0 < SEQ; chunk0 += KCHUNK) {
        // ---- Phase A: scores (each wave: 64 keys = 4 key-tiles) ----
#pragma unroll
        for (int t4 = 0; t4 < 4; ++t4) {
            int kt0 = chunk0 + wave * 64 + t4 * 16;
            v8f st = (v8f)0.0f;
#pragma unroll
            for (int c = 0; c < 4; ++c) {
                const __bf16* krow = kb + (size_t)(kt0 + lr) * DIM + h * HD +
                                     c * 32 + hi2 * 16;
                if (chunk0 + KCHUNK < SEQ)
                    __builtin_prefetch(krow + (size_t)KCHUNK * DIM, 0, 1);
                v8bf lo = *(const v8bf*)krow;
                v8bf hb = *(const v8bf*)(krow + 8);
                st = wmma_bf16(qf[c], cat8(lo, hb), st);
            }
            int key = kt0 + lr;
            float mv = (key >= slen) ? -1e30f : 0.f;
            int colBase = wave * 64 + t4 * 16 + lr;
#pragma unroll
            for (int r = 0; r < 8; ++r)
                sc[(r + hi2 * 8) * SCS + colBase] = st[r] * scale + mv;
        }
        __syncthreads();

        // ---- Phase B: online softmax over the chunk ----
        {
            float* srow = &sc[srow_id * SCS];
            float cmax = -3.0e38f;
            for (int j = sl; j < KCHUNK; j += 16) cmax = fmaxf(cmax, srow[j]);
#pragma unroll
            for (int off = 8; off > 0; off >>= 1)
                cmax = fmaxf(cmax, __shfl_xor(cmax, off, 32));
            float m_new = fmaxf(m_run, cmax);
            float f = __expf(m_run - m_new);
            float psum = 0.f;
            for (int j = sl; j < KCHUNK; j += 16) {
                float e = __expf(srow[j] - m_new);
                srow[j] = e;
                psum += e;
            }
#pragma unroll
            for (int off = 8; off > 0; off >>= 1)
                psum += __shfl_xor(psum, off, 32);
            s_run = s_run * f + psum;
            m_run = m_new;
            if (sl == 0) fac[srow_id] = f;
        }
        __syncthreads();

        // ---- Phase C: O += P @ V (wave owns dims [wave*16, wave*16+16)) ----
#pragma unroll
        for (int r = 0; r < 8; ++r) oacc[r] *= fac[r + hi2 * 8];

        stageV(0, chunk0);
        wait_async0();
        __syncthreads();
        for (int t4 = 0; t4 < 16; ++t4) {          // 16 tiles of 32 keys
            if (t4 + 1 < 16) stageV((t4 + 1) & 1, chunk0 + (t4 + 1) * 32);

            const float* pr = &sc[lr * SCS + t4 * 32];
            float4 a0 = *(const float4*)(pr + hi2 * 8);
            float4 a1 = *(const float4*)(pr + hi2 * 8 + 4);
            float4 a2 = *(const float4*)(pr + 16 + hi2 * 8);
            float4 a3 = *(const float4*)(pr + 16 + hi2 * 8 + 4);
            v16bf pf;
            pf[0]=(__bf16)a0.x; pf[1]=(__bf16)a0.y; pf[2]=(__bf16)a0.z; pf[3]=(__bf16)a0.w;
            pf[4]=(__bf16)a1.x; pf[5]=(__bf16)a1.y; pf[6]=(__bf16)a1.z; pf[7]=(__bf16)a1.w;
            pf[8]=(__bf16)a2.x; pf[9]=(__bf16)a2.y; pf[10]=(__bf16)a2.z; pf[11]=(__bf16)a2.w;
            pf[12]=(__bf16)a3.x; pf[13]=(__bf16)a3.y; pf[14]=(__bf16)a3.z; pf[15]=(__bf16)a3.w;

            const __bf16* vrow = &vT[t4 & 1][(wave * 16 + lr) * VLD];
            v8bf lo = *(const v8bf*)(vrow + hi2 * 8);
            v8bf hb = *(const v8bf*)(vrow + 16 + hi2 * 8);
            oacc = wmma_bf16(pf, cat8(lo, hb), oacc);

            wait_async0();
            __syncthreads();
        }
    }

    if (sl == 0) inv_s[srow_id] = 1.0f / s_run;
    __syncthreads();
#pragma unroll
    for (int r = 0; r < 8; ++r) {
        int M = r + hi2 * 8;
        float o = oacc[r] * inv_s[M];
        attnb[(size_t)(qbase + M) * DIM + h * HD + wave * 16 + lr] = (__bf16)o;
    }
}

// ---------------------------------------------------------------- launch
extern "C" void kernel_launch(void* const* d_in, const int* in_sizes, int n_in,
                              void* d_out, int out_size, void* d_ws, size_t ws_size,
                              hipStream_t stream) {
    const float* x        = (const float*)d_in[0];
    const int*   seq_lens = (const int*)  d_in[1];
    /* d_in[2]: grid_sizes (3,32,32) — layout hardcoded */
    const float* freqs    = (const float*)d_in[3];
    const float* Wq = (const float*)d_in[4];  const float* bq = (const float*)d_in[5];
    const float* Wk = (const float*)d_in[6];  const float* bk = (const float*)d_in[7];
    const float* Wv = (const float*)d_in[8];  const float* bv = (const float*)d_in[9];
    const float* Wo = (const float*)d_in[10]; const float* bo = (const float*)d_in[11];
    const float* gq = (const float*)d_in[12]; const float* gk = (const float*)d_in[13];
    float* out = (float*)d_out;

    const size_t nX = (size_t)SEQ * DIM;     // 4,718,592
    const size_t nW = (size_t)DIM * DIM;     // 2,359,296
    char* p = (char*)d_ws;
    __bf16* xb   = (__bf16*)p; p += nX * 2;
    __bf16* Wqt  = (__bf16*)p; p += nW * 2;  // W^T bf16 [N][K]
    __bf16* Wkt  = (__bf16*)p; p += nW * 2;
    __bf16* Wvt  = (__bf16*)p; p += nW * 2;
    __bf16* Wot  = (__bf16*)p; p += nW * 2;
    float*  q0f  = (float*)p;  p += nX * 4;
    float*  k0f  = (float*)p;  p += nX * 4;
    float*  v0f  = (float*)p;  p += nX * 4;
    __bf16* qbn  = (__bf16*)p; p += nX * 2;
    __bf16* kbn  = (__bf16*)p; p += nX * 2;
    __bf16* vt   = (__bf16*)p; p += nX * 2;  // V^T bf16 [DIM][SEQ]
    __bf16* attb = (__bf16*)p; p += nX * 2;

    cvt_f32_bf16<<<(int)((nX + 255) / 256), 256, 0, stream>>>(x, xb, (int)nX);
    dim3 tgridW(DIM / 32, DIM / 32);    // (48, 48)
    cvt_transpose<<<tgridW, 256, 0, stream>>>(Wq, Wqt, DIM, DIM);
    cvt_transpose<<<tgridW, 256, 0, stream>>>(Wk, Wkt, DIM, DIM);
    cvt_transpose<<<tgridW, 256, 0, stream>>>(Wv, Wvt, DIM, DIM);
    cvt_transpose<<<tgridW, 256, 0, stream>>>(Wo, Wot, DIM, DIM);

    dim3 ggrid(DIM / BN, SEQ / BM);     // (12, 24)
    gemm_bf16<<<ggrid, 256, 0, stream>>>(xb, Wqt, bq, q0f, SEQ, DIM, DIM);
    gemm_bf16<<<ggrid, 256, 0, stream>>>(xb, Wkt, bk, k0f, SEQ, DIM, DIM);
    gemm_bf16<<<ggrid, 256, 0, stream>>>(xb, Wvt, bv, v0f, SEQ, DIM, DIM);

    rms_rope<<<SEQ, 256, 0, stream>>>(q0f, k0f, freqs, gq, gk, qbn, kbn);

    dim3 tgridV(DIM / 32, SEQ / 32);    // (48, 96): v0f [SEQ][DIM] -> vt [DIM][SEQ]
    cvt_transpose<<<tgridV, 256, 0, stream>>>(v0f, vt, SEQ, DIM);

    flash_attn<<<dim3(SEQ / 16, NH), 256, 0, stream>>>(qbn, kbn, vt, seq_lens, attb);

    gemm_bf16<<<ggrid, 256, 0, stream>>>(attb, Wot, bo, out, SEQ, DIM, DIM);
}
